// LinearAttention_42262478193306
// MI455X (gfx1250) — matrix operand order repaired and checked by reading, not compiled, not run on hardware
//
#include <hip/hip_runtime.h>

typedef __attribute__((ext_vector_type(16))) _Float16 v16h;
typedef __attribute__((ext_vector_type(8)))  _Float16 v8h;
typedef __attribute__((ext_vector_type(8)))  float    v8f;

#define B_   4
#define T_   4096
#define D_   1024
#define H_   16
#define HD_  64
#define M_   (B_*T_)     // 16384
#define N3_  (3*D_)      // 3072

#if defined(__gfx1250__) && __has_builtin(__builtin_amdgcn_global_load_async_to_lds_b128)
#define ASYNC_LDS 1
#else
#define ASYNC_LDS 0
#endif

#if ASYNC_LDS
typedef __attribute__((ext_vector_type(4))) int v4i_;
typedef __attribute__((address_space(1))) v4i_* gptr_t;  // global int4*
typedef __attribute__((address_space(3))) v4i_* lptr_t;  // LDS int4*
#endif

__device__ __forceinline__ v16h make_v16(v8h lo, v8h hi) {
  v16h r;
#pragma unroll
  for (int i = 0; i < 8; ++i) { r[i] = lo[i]; r[i + 8] = hi[i]; }
  return r;
}

// stage one thread's 32B stripe of a tile: global -> LDS
__device__ __forceinline__ void stage32B(const _Float16* __restrict__ g,
                                         _Float16* s) {
#if ASYNC_LDS
  gptr_t gp = (gptr_t)g;
  lptr_t lp = (lptr_t)s;
  // INST_OFFSET is added to both the global and LDS addresses (ISA 08 §4.4)
  __builtin_amdgcn_global_load_async_to_lds_b128(gp, lp, 0, 0);
  __builtin_amdgcn_global_load_async_to_lds_b128(gp, lp, 16, 0);
#else
  *(v16h*)s = *(const v16h*)g;
#endif
}

__device__ __forceinline__ void wait_async() {
#if ASYNC_LDS
#if __has_builtin(__builtin_amdgcn_s_wait_asynccnt)
  __builtin_amdgcn_s_wait_asynccnt(0);
#else
  asm volatile("s_wait_asynccnt 0" ::: "memory");
#endif
#endif
}

// ---------------------------------------------------------------- conversion
__global__ void cvt_f32_f16(const float* __restrict__ in,
                            _Float16* __restrict__ out, int n) {
  int i = blockIdx.x * blockDim.x + threadIdx.x;
  int stride = gridDim.x * blockDim.x;
  for (; i < n; i += stride) out[i] = (_Float16)in[i];
}

// ---------------------------------------------------------------- big GEMM
// C[M,N] = A[M,K] * W[N,K]^T  (f16 in, f32 accum via v_wmma_f32_16x16x32_f16)
// epi==0: feature map (elu+1) on cols < 2*D_, store f16 to out16 (stride ldo)
// epi==1: add bias[n], store f32 to out32 (stride ldo)
__global__ __launch_bounds__(256) void wmma_gemm(
    const _Float16* __restrict__ A, const _Float16* __restrict__ W,
    int K, int epi,
    _Float16* __restrict__ out16, float* __restrict__ out32,
    const float* __restrict__ bias, int ldo) {
  __shared__ __align__(32) _Float16 Alds[2][128 * 32];
  __shared__ __align__(32) _Float16 Blds[2][128 * 32];  // stored [n][k]

  const int tid  = threadIdx.x;
  const int lane = tid & 31, wid = tid >> 5;
  const int sel  = lane >> 4, lr = lane & 15;
  const int waveM = wid & 1;    // 2 waves down   -> 64 rows each
  const int waveN = wid >> 1;   // 4 waves across -> 32 cols each
  const int rowBlk = blockIdx.y * 128;
  const int colBlk = blockIdx.x * 128;

  const int ldRow  = tid >> 1;         // 0..127
  const int ldPart = (tid & 1) * 16;   // halves
  const int sOff   = ldRow * 32 + ldPart;

  const _Float16* gA = A + (size_t)(rowBlk + ldRow) * K + ldPart;
  const _Float16* gB = W + (size_t)(colBlk + ldRow) * K + ldPart;

  v8f zero = {};
  v8f acc[4][2];
#pragma unroll
  for (int i = 0; i < 4; ++i)
#pragma unroll
    for (int j = 0; j < 2; ++j) acc[i][j] = zero;

  // prologue: stage slice 0 into buffer 0
  stage32B(gA, &Alds[0][sOff]);
  stage32B(gB, &Blds[0][sOff]);
#if !ASYNC_LDS
  __builtin_prefetch(gA + 32, 0, 1);
  __builtin_prefetch(gB + 32, 0, 1);
#endif
  wait_async();
  __syncthreads();

  const int NK = K >> 5;
  for (int it = 0; it < NK; ++it) {
    const int cur = it & 1;
    if (it + 1 < NK) {  // issue next slice into the other buffer (overlapped)
      stage32B(gA + (it + 1) * 32, &Alds[cur ^ 1][sOff]);
      stage32B(gB + (it + 1) * 32, &Blds[cur ^ 1][sOff]);
    }

    v16h af[4];
#pragma unroll
    for (int mt = 0; mt < 4; ++mt) {
      const _Float16* ar = &Alds[cur][(waveM * 64 + mt * 16 + lr) * 32];
      v8h lo = *(const v8h*)(ar + sel * 8);
      v8h hi = *(const v8h*)(ar + 16 + sel * 8);
      af[mt] = make_v16(lo, hi);
    }
    v16h bf[2];
#pragma unroll
    for (int nt = 0; nt < 2; ++nt)
      bf[nt] = make_v16(*(const v8h*)(&Blds[cur][(waveN * 32 + nt * 16 + lr) * 32] + sel * 8), *(const v8h*)(&Blds[cur][(waveN * 32 + nt * 16 + lr) * 32] + 16 + sel * 8));

#pragma unroll
    for (int mt = 0; mt < 4; ++mt)
#pragma unroll
      for (int nt = 0; nt < 2; ++nt)
        acc[mt][nt] = __builtin_amdgcn_wmma_f32_16x16x32_f16(
            false, af[mt], false, bf[nt], (short)0, acc[mt][nt], false, false);

    wait_async();      // next buffer's async loads have landed
    __syncthreads();   // all waves done reading cur / writing next
  }

#pragma unroll
  for (int mt = 0; mt < 4; ++mt) {
#pragma unroll
    for (int nt = 0; nt < 2; ++nt) {
      int row0 = rowBlk + waveM * 64 + mt * 16 + sel * 8;
      int col  = colBlk + waveN * 32 + nt * 16 + lr;
      if (epi == 0) {
        bool feat = col < 2 * D_;     // Q and K get elu(z)+1
        _Float16* dst = out16 + (size_t)row0 * ldo + col;
#pragma unroll
        for (int r = 0; r < 8; ++r) {
          float v = acc[mt][nt][r];
          if (feat) v = (v > 0.0f) ? (v + 1.0f) : __expf(v);
          dst[(size_t)r * ldo] = (_Float16)v;
        }
      } else {
        float bv = bias[col];
        float* dst = out32 + (size_t)row0 * ldo + col;
#pragma unroll
        for (int r = 0; r < 8; ++r)
          dst[(size_t)r * ldo] = acc[mt][nt][r] + bv;
      }
    }
  }
}

// ---------------------------------------------------------------- K column sums
__global__ void ksum_kernel(const _Float16* __restrict__ qkv,
                            float* __restrict__ ksum) {
  int bh = blockIdx.x, d = threadIdx.x;
  int b = bh >> 4, h = bh & 15;
  const _Float16* kp = qkv + (size_t)b * T_ * N3_ + D_ + h * HD_ + d;
  float s = 0.0f;
  for (int t = 0; t < T_; ++t) s += (float)kp[(size_t)t * N3_];
  ksum[bh * HD_ + d] = s;
}

// ---------------------------------------------------------------- KV = K^T V per head
__global__ __launch_bounds__(512) void kv_kernel(const _Float16* __restrict__ qkv,
                                                 float* __restrict__ kvg) {
  __shared__ __align__(32) _Float16 Klds[64 * 32];  // [d][t]
  __shared__ __align__(32) _Float16 Vlds[64 * 32];  // [e][t]
  int bh = blockIdx.x, b = bh >> 4, h = bh & 15;
  int tid = threadIdx.x, lane = tid & 31, wid = tid >> 5;
  int sel = lane >> 4, lr = lane & 15;
  int dtile = wid >> 2, etile = wid & 3;            // 4x4 tiles of 16x16

  int st = tid >> 4;              // t within chunk (0..31)
  int d0 = (tid & 15) * 4;        // 4 features per thread

  v8f acc = {};
  for (int t0 = 0; t0 < T_; t0 += 32) {
    size_t row = (size_t)(b * T_ + t0 + st) * N3_;
    const _Float16* kp = qkv + row + D_ + h * HD_ + d0;
    const _Float16* vp = qkv + row + 2 * D_ + h * HD_ + d0;
#pragma unroll
    for (int j = 0; j < 4; ++j) {
      Klds[(d0 + j) * 32 + st] = kp[j];
      Vlds[(d0 + j) * 32 + st] = vp[j];
    }
    __syncthreads();
    const _Float16* ar = &Klds[(dtile * 16 + lr) * 32];
    v8h lo = *(const v8h*)(ar + sel * 8);
    v8h hi = *(const v8h*)(ar + 16 + sel * 8);
    v16h af = make_v16(lo, hi);
    v16h bf = make_v16(*(const v8h*)(&Vlds[(etile * 16 + lr) * 32] + sel * 8), *(const v8h*)(&Vlds[(etile * 16 + lr) * 32] + 16 + sel * 8));
    acc = __builtin_amdgcn_wmma_f32_16x16x32_f16(false, af, false, bf,
                                                 (short)0, acc, false, false);
    __syncthreads();
  }
  int drow = dtile * 16 + sel * 8;
  int e    = etile * 16 + lr;
#pragma unroll
  for (int r = 0; r < 8; ++r)
    kvg[((size_t)bh * 64 + drow + r) * 64 + e] = acc[r];
}

// ---------------------------------------------------------------- context = (Q KV) / Z
__global__ __launch_bounds__(256) void ctx_kernel(
    const _Float16* __restrict__ qkv, const float* __restrict__ kvg,
    const float* __restrict__ ksum, _Float16* __restrict__ ctx) {
  __shared__ __align__(32) _Float16 Bl[64 * 64];   // [e][d] = KV[d][e]
  __shared__ float Zl[128];
  __shared__ float Ks[64];
  int mblk = blockIdx.x, bh = blockIdx.y;
  int b = bh >> 4, h = bh & 15;
  int tid = threadIdx.x, lane = tid & 31, wid = tid >> 5;
  int sel = lane >> 4, lr = lane & 15;
  int t0 = mblk * 128;

  {
    const float* src = kvg + (size_t)bh * 4096;
#pragma unroll
    for (int i = 0; i < 16; ++i) {
      int idx = tid * 16 + i;
      int d = idx >> 6, e = idx & 63;
      Bl[e * 64 + d] = (_Float16)src[idx];
    }
  }
  if (tid < 64) Ks[tid] = ksum[bh * 64 + tid];
  __syncthreads();
  if (tid < 128) {
    const _Float16* q = qkv + (size_t)(b * T_ + t0 + tid) * N3_ + h * HD_;
    float z = 1e-6f;
#pragma unroll
    for (int d = 0; d < 64; ++d) z += (float)q[d] * Ks[d];
    Zl[tid] = z;
  }
  __syncthreads();

  v8f zero = {};
  v8f acc[4];
#pragma unroll
  for (int nt = 0; nt < 4; ++nt) acc[nt] = zero;

  int m = t0 + wid * 16 + lr;  // this lane's A row
  const _Float16* qrow = qkv + (size_t)(b * T_ + m) * N3_ + h * HD_;
#pragma unroll
  for (int kc = 0; kc < 2; ++kc) {
    const _Float16* p = qrow + kc * 32;
    v8h lo = *(const v8h*)(p + sel * 8);
    v8h hi = *(const v8h*)(p + 16 + sel * 8);
    v16h af = make_v16(lo, hi);
#pragma unroll
    for (int nt = 0; nt < 4; ++nt) {
      v16h bf = make_v16(*(const v8h*)(&Bl[(nt * 16 + lr) * 64 + kc * 32] + sel * 8), *(const v8h*)(&Bl[(nt * 16 + lr) * 64 + kc * 32] + 16 + sel * 8));
      acc[nt] = __builtin_amdgcn_wmma_f32_16x16x32_f16(false, af, false, bf,
                                                       (short)0, acc[nt], false, false);
    }
  }
  int rloc = wid * 16 + sel * 8;
#pragma unroll
  for (int nt = 0; nt < 4; ++nt) {
    int col = h * HD_ + nt * 16 + lr;
#pragma unroll
    for (int r = 0; r < 8; ++r) {
      float v = acc[nt][r] / Zl[rloc + r];
      ctx[(size_t)(b * T_ + t0 + rloc + r) * D_ + col] = (_Float16)v;
    }
  }
}

// ---------------------------------------------------------------- launch
extern "C" void kernel_launch(void* const* d_in, const int* in_sizes, int n_in,
                              void* d_out, int out_size, void* d_ws, size_t ws_size,
                              hipStream_t stream) {
  (void)in_sizes; (void)n_in; (void)out_size; (void)ws_size;
  const float* x  = (const float*)d_in[0];
  const float* Wq = (const float*)d_in[1];
  const float* Wk = (const float*)d_in[2];
  const float* Wv = (const float*)d_in[3];
  const float* Wo = (const float*)d_in[4];
  const float* bo = (const float*)d_in[5];

  char* w = (char*)d_ws;
  _Float16* xf   = (_Float16*)w;  w += (size_t)M_ * D_ * 2;     // x in f16
  _Float16* wcat = (_Float16*)w;  w += (size_t)3 * D_ * D_ * 2; // [Wq;Wk;Wv] f16 [3072,1024]
  _Float16* wof  = (_Float16*)w;  w += (size_t)D_ * D_ * 2;     // Wo f16 [1024,1024]
  _Float16* qkv  = (_Float16*)w;  w += (size_t)M_ * N3_ * 2;    // fused Q|K|V  [M,3072]
  float*    kvg  = (float*)w;     w += (size_t)B_ * H_ * HD_ * HD_ * 4;
  float*    ks   = (float*)w;     w += (size_t)B_ * H_ * HD_ * 4;
  _Float16* ctx  = (_Float16*)w;                                 // [M,1024] f16

  cvt_f32_f16<<<2048, 256, 0, stream>>>(x, xf, M_ * D_);
  cvt_f32_f16<<<512, 256, 0, stream>>>(Wq, wcat, D_ * D_);
  cvt_f32_f16<<<512, 256, 0, stream>>>(Wk, wcat + (size_t)D_ * D_, D_ * D_);
  cvt_f32_f16<<<512, 256, 0, stream>>>(Wv, wcat + (size_t)2 * D_ * D_, D_ * D_);
  cvt_f32_f16<<<512, 256, 0, stream>>>(Wo, wof, D_ * D_);

  // fused QKV projection + feature map (epi 0)
  wmma_gemm<<<dim3(N3_ / 128, M_ / 128), 256, 0, stream>>>(
      xf, wcat, D_, 0, qkv, nullptr, nullptr, N3_);

  ksum_kernel<<<B_ * H_, HD_, 0, stream>>>(qkv, ks);
  kv_kernel<<<B_ * H_, 512, 0, stream>>>(qkv, kvg);
  ctx_kernel<<<dim3(T_ / 128, B_ * H_), 256, 0, stream>>>(qkv, kvg, ks, ctx);

  // output projection + bias (epi 1) straight into d_out
  wmma_gemm<<<dim3(D_ / 128, M_ / 128), 256, 0, stream>>>(
      ctx, wof, D_, 1, nullptr, (float*)d_out, bo, D_);
}
